// Fp8Linear_23278722744748
// MI455X (gfx1250) — 8-bit matrix operand order repaired and checked by reading, compiled, not run on hardware
//
#include <hip/hip_runtime.h>
#include <hip/hip_bf16.h>
#include <stdint.h>

constexpr int M  = 4096;
constexpr int K  = 4096;
constexpr int N  = 14336;
constexpr int GRP = 128;
constexpr int KG  = K / GRP;
constexpr float FP8_MAX = 448.0f;
constexpr float QEPS    = 1e-4f;

constexpr int LDS_STRIDE = 144;
constexpr int OUT_STRIDE = 136;

typedef __attribute__((ext_vector_type(16))) int   v16i;
typedef __attribute__((ext_vector_type(8)))  float v8f;

__device__ __forceinline__ void async_ld_b128(uint32_t lds_off, const uint8_t* g) {
  asm volatile("global_load_async_to_lds_b128 %0, %1, off"
               :: "v"(lds_off), "v"(g) : "memory");
}
template <int N_>
__device__ __forceinline__ void wait_async() {
  asm volatile("s_wait_asynccnt %0" :: "i"(N_) : "memory");
}

__device__ __forceinline__ unsigned f32_to_e4m3(float x) {
  unsigned s = (__float_as_uint(x) >> 31) << 7;
  float a = fminf(fabsf(x), FP8_MAX);
  if (a < 0.015625f) {
    unsigned q = (unsigned)(int)rintf(a * 512.0f);
    return s | q;
  }
  unsigned ua   = __float_as_uint(a);
  unsigned mant = ua & 0x7FFFFFu;
  int      e    = (int)(ua >> 23) - 127;
  unsigned keep = mant >> 20;
  unsigned rest = mant & 0xFFFFFu;
  keep += (rest > 0x80000u) || ((rest == 0x80000u) && (keep & 1u));
  if (keep == 8u) { keep = 0u; ++e; }
  if (e > 8) { e = 8; keep = 6u; }
  return s | ((unsigned)(e + 7) << 3) | keep;
}

__global__ __launch_bounds__(256) void quant_w_kernel(
    const float* __restrict__ w, uint8_t* __restrict__ wq) {
  size_t i = ((size_t)blockIdx.x * 256 + threadIdx.x) * 4;
  float4 v = *(const float4*)(w + i);
  unsigned p = f32_to_e4m3(v.x)
             | (f32_to_e4m3(v.y) << 8)
             | (f32_to_e4m3(v.z) << 16)
             | (f32_to_e4m3(v.w) << 24);
  *(unsigned*)(wq + i) = p;
}

__global__ __launch_bounds__(256) void quant_x_kernel(
    const __hip_bfloat16* __restrict__ x, uint8_t* __restrict__ xq,
    float* __restrict__ sa_t) {
  const int lane = threadIdx.x & 31;
  const int gid  = blockIdx.x * 8 + (threadIdx.x >> 5);
  const int m = gid >> 5;
  const int g = gid & 31;

  const size_t base = (size_t)m * K + (size_t)g * GRP + lane * 4;
  uint2 rw = *(const uint2*)((const uint16_t*)x + base);
  float f0 = __uint_as_float(rw.x << 16);
  float f1 = __uint_as_float(rw.x & 0xFFFF0000u);
  float f2 = __uint_as_float(rw.y << 16);
  float f3 = __uint_as_float(rw.y & 0xFFFF0000u);

  float amax = fmaxf(fmaxf(fabsf(f0), fabsf(f1)), fmaxf(fabsf(f2), fabsf(f3)));
  #pragma unroll
  for (int off = 16; off > 0; off >>= 1)
    amax = fmaxf(amax, __shfl_xor(amax, off, 32));

  const float scale = fmaxf(amax, QEPS) * (1.0f / FP8_MAX);
  const float inv   = FP8_MAX / fmaxf(amax, QEPS);

  unsigned p = f32_to_e4m3(f0 * inv)
             | (f32_to_e4m3(f1 * inv) << 8)
             | (f32_to_e4m3(f2 * inv) << 16)
             | (f32_to_e4m3(f3 * inv) << 24);
  *(unsigned*)(xq + base) = p;
  if (lane == 0) sa_t[(size_t)g * M + m] = scale;
}

__global__ __launch_bounds__(256) void fp8_gemm_kernel(
    const uint8_t* __restrict__ xq, const float* __restrict__ sa_t,
    const uint8_t* __restrict__ wq, const float* __restrict__ sw,
    const float* __restrict__ bias, __hip_bfloat16* __restrict__ out) {
  __shared__ uint8_t sA[2][128 * LDS_STRIDE];
  __shared__ uint8_t sB[2][128 * LDS_STRIDE];

  const int tid  = threadIdx.x;
  const int lane = tid & 31;
  const int wave = tid >> 5;
  const int wm   = wave >> 1;
  const int wn   = wave & 1;
  const int l15  = lane & 15;
  const int hi   = lane >> 4;

  const int bm = blockIdx.y * 128;
  const int bn = blockIdx.x * 128;
  const int m0 = bm + wm * 32;
  const int n0 = bn + wn * 64;
  const int nb = bn >> 7;

  const int      crow = tid >> 3;
  const int      cc   = (tid & 7) * 16;
  const uint8_t* gA0  = xq + (size_t)(bm + crow) * K + cc;
  const uint8_t* gB0  = wq + (size_t)(bn + crow) * K + cc;

  auto issue_stage = [&](int g, int buf) {
    const uint8_t* ga = gA0 + (size_t)g * GRP;
    const uint8_t* gb = gB0 + (size_t)g * GRP;
    #pragma unroll
    for (int i = 0; i < 4; ++i) {
      uint32_t la = (uint32_t)(uintptr_t)&sA[buf][(crow + i * 32) * LDS_STRIDE + cc];
      async_ld_b128(la, ga + (size_t)(i * 32) * K);
    }
    #pragma unroll
    for (int i = 0; i < 4; ++i) {
      uint32_t lb = (uint32_t)(uintptr_t)&sB[buf][(crow + i * 32) * LDS_STRIDE + cc];
      async_ld_b128(lb, gb + (size_t)(i * 32) * K);
    }
  };

  const v8f zero = {0.f, 0.f, 0.f, 0.f, 0.f, 0.f, 0.f, 0.f};
  v8f acc[2][4];
  #pragma unroll
  for (int mi = 0; mi < 2; ++mi)
    #pragma unroll
    for (int ni = 0; ni < 4; ++ni) acc[mi][ni] = zero;

  issue_stage(0, 0);

  for (int g = 0; g < KG; ++g) {
    const int cur = g & 1;
    if (g + 1 < KG) {
      issue_stage(g + 1, cur ^ 1);
      wait_async<8>();
    } else {
      wait_async<0>();
    }
    __syncthreads();

    const float  swg = sw[(size_t)nb * KG + g];
    const float* sp  = sa_t + (size_t)g * M + m0 + hi * 8;
    float4 s00 = *(const float4*)(sp);
    float4 s01 = *(const float4*)(sp + 4);
    float4 s10 = *(const float4*)(sp + 16);
    float4 s11 = *(const float4*)(sp + 20);
    float scl[2][8] = {
      {s00.x * swg, s00.y * swg, s00.z * swg, s00.w * swg,
       s01.x * swg, s01.y * swg, s01.z * swg, s01.w * swg},
      {s10.x * swg, s10.y * swg, s10.z * swg, s10.w * swg,
       s11.x * swg, s11.y * swg, s11.z * swg, s11.w * swg}};

    const uint8_t* As = &sA[cur][0];
    const uint8_t* Bs = &sB[cur][0];

    v16i af[2];
    #pragma unroll
    for (int mi = 0; mi < 2; ++mi) {
      const uint8_t* ap = As + (wm * 32 + mi * 16 + l15) * LDS_STRIDE + hi * 8;
      #pragma unroll
      for (int j = 0; j < 8; ++j) {
        uint2 t = *(const uint2*)(ap + hi * 8 + (j >> 1) * 32 + (j & 1) * 8);
        af[mi][2 * j]     = (int)t.x;
        af[mi][2 * j + 1] = (int)t.y;
      }
    }

    v16i bfr[4];
    #pragma unroll
    for (int ni = 0; ni < 4; ++ni) {
      const uint8_t* bp = Bs + (wn * 64 + ni * 16 + l15) * LDS_STRIDE + hi * 16;
      #pragma unroll
      for (int q = 0; q < 4; ++q) {
        uint4 t = *(const uint4*)(bp + q * 32);
        bfr[ni][4 * q + 0] = (int)t.x;
        bfr[ni][4 * q + 1] = (int)t.y;
        bfr[ni][4 * q + 2] = (int)t.z;
        bfr[ni][4 * q + 3] = (int)t.w;
      }
    }

    #pragma unroll
    for (int mi = 0; mi < 2; ++mi)
      #pragma unroll
      for (int ni = 0; ni < 4; ++ni) {
        v8f part = __builtin_amdgcn_wmma_f32_16x16x128_fp8_fp8(
            af[mi], bfr[ni], (short)0, zero, false, false);
        #pragma unroll
        for (int r = 0; r < 8; ++r)
          acc[mi][ni][r] += part[r] * scl[mi][r];
      }

    __syncthreads();
  }

  __hip_bfloat16* ot = (__hip_bfloat16*)&sA[0][0];

  #pragma unroll
  for (int ni = 0; ni < 4; ++ni) {
    const int   lcol = wn * 64 + ni * 16 + l15;
    const float bv   = bias[bn + lcol];
    #pragma unroll
    for (int mi = 0; mi < 2; ++mi) {
      #pragma unroll
      for (int r = 0; r < 8; ++r) {
        const int lrow = wm * 32 + mi * 16 + hi * 8 + r;
        ot[lrow * OUT_STRIDE + lcol] = __float2bfloat16(acc[mi][ni][r] + bv);
      }
    }
  }
  __syncthreads();

  const int orow = tid >> 1;
  const int oh   = tid & 1;
  const __hip_bfloat16* src = ot + orow * OUT_STRIDE + oh * 64;
  __hip_bfloat16*       dst = out + (size_t)(bm + orow) * N + bn + oh * 64;
  #pragma unroll
  for (int j = 0; j < 8; ++j) {
    uint4 t = *(const uint4*)(src + j * 8);
    *(uint4*)(dst + j * 8) = t;
  }
}

extern "C" void kernel_launch(void* const* d_in, const int* in_sizes, int n_in,
                              void* d_out, int out_size, void* d_ws, size_t ws_size,
                              hipStream_t stream) {
  const __hip_bfloat16* x  = (const __hip_bfloat16*)d_in[0];
  const float* w    = (const float*)d_in[1];
  const float* wscl = (const float*)d_in[2];
  const float* bias = (const float*)d_in[3];
  __hip_bfloat16* out = (__hip_bfloat16*)d_out;

  uint8_t* xq = (uint8_t*)d_ws;
  float*   sa = (float*)((uint8_t*)d_ws + (size_t)M * K);
  uint8_t* wq = (uint8_t*)d_ws + (size_t)M * K + (size_t)KG * M * 4;

  quant_w_kernel<<<(unsigned)(((size_t)N * K) / 1024), 256, 0, stream>>>(w, wq);
  quant_x_kernel<<<(M * KG) / 8, 256, 0, stream>>>(x, xq, sa);

  dim3 grid(N / 128, M / 128);
  fp8_gemm_kernel<<<grid, 256, 0, stream>>>(xq, sa, wq, wscl, bias, out);
}
